// SparseLinear_74698071212292
// MI455X (gfx1250) — matrix operand order repaired and checked by reading, not compiled, not run on hardware
//
#include <hip/hip_runtime.h>
#include <hip/hip_bf16.h>

// ---- types ----
typedef __attribute__((ext_vector_type(16))) __bf16 v16bf;
typedef __attribute__((ext_vector_type(8)))  __bf16 v8bf;
typedef __attribute__((ext_vector_type(4)))  __bf16 v4bf;
typedef __attribute__((ext_vector_type(8)))  float  v8f;
typedef __attribute__((ext_vector_type(4)))  float  v4f;
typedef __attribute__((ext_vector_type(4)))  int    v4i;

#define IN_F   4096
#define OUT_F  4096
#define BATCH  8192

// fast-path tiling
#define BM 128
#define BN 256
#define BK 64
#define KST 72          // LDS row stride in bf16 (144B = 16B-aligned, bank-friendly)

// fallback tiling (round-1 kernel)
#define FBM 128
#define FBN 128
#define FBK 32
#define FSTRIDE 40

// ---------------- async copy helpers ----------------
// Builtin prototype (per hipcc diagnostic): (v4i AS1*, v4i AS3*, imm offset, imm cpol)
typedef __attribute__((address_space(1))) v4i* gv4i_t;
typedef __attribute__((address_space(3))) v4i* lv4i_t;

__device__ __forceinline__ void async_copy16(const __bf16* g, __bf16* l) {
#if __has_builtin(__builtin_amdgcn_global_load_async_to_lds_b128)
    __builtin_amdgcn_global_load_async_to_lds_b128(
        (gv4i_t)(void*)g, (lv4i_t)(void*)l, 0, 0);
#else
    *(v8bf*)l = *(const v8bf*)g;   // sync fallback
#endif
}

__device__ __forceinline__ void wait_async0() {
#if __has_builtin(__builtin_amdgcn_s_wait_asynccnt)
    __builtin_amdgcn_s_wait_asynccnt(0);
#elif __has_builtin(__builtin_amdgcn_global_load_async_to_lds_b128)
    asm volatile("s_wait_asynccnt 0" ::: "memory");
#endif
}

// ---------------- zero the dense W workspace ----------------
__global__ __launch_bounds__(256) void zero_ws_kernel(v4f* __restrict__ p) {
    size_t i = (size_t)blockIdx.x * 256 + threadIdx.x;
    v4f z = {0.f, 0.f, 0.f, 0.f};
    p[i] = z;
}

// ---------------- scatter COO -> dense W [out][in] fp32 ----------------
__global__ __launch_bounds__(256) void scatter_coo_kernel(
    const int* __restrict__ rows, const int* __restrict__ cols,
    const float* __restrict__ vals, float* __restrict__ W, int nnz) {
    int i = blockIdx.x * 256 + threadIdx.x;
    if (i < nnz) {
        atomicAdd(&W[(size_t)rows[i] * IN_F + cols[i]], vals[i]);
    }
}

// ---------------- fp32 -> bf16 bulk convert (4 elems/thread) ----------------
__global__ __launch_bounds__(256) void cvt_bf16_kernel(
    const float* __restrict__ src, __bf16* __restrict__ dst) {
    size_t i = (size_t)blockIdx.x * 256 + threadIdx.x;
    v4f v = *(const v4f*)(src + i * 4);
    *(v4bf*)(dst + i * 4) = __builtin_convertvector(v, v4bf);
}

// =====================================================================
// Fast path: Y = Xh * Wh^T + bias, bf16 operands from global, f32 acc.
// 128x256 block tile, 8 waves (2x4), wave tile 64x64, BK=64,
// double-buffered LDS filled with async global->LDS copies.
// =====================================================================
__global__ __launch_bounds__(256, 1) void gemm_bf16_async_kernel(
    const __bf16* __restrict__ Xh,   // [BATCH][IN_F]
    const __bf16* __restrict__ Wh,   // [OUT_F][IN_F]
    const float*  __restrict__ bias, // [OUT_F]
    float* __restrict__ Y)           // [BATCH][OUT_F]
{
    __shared__ __align__(16) __bf16 Asb[2][BM][KST];
    __shared__ __align__(16) __bf16 Bsb[2][BN][KST];

    const int tid  = threadIdx.x;
    const int lane = tid & 31;
    const int wave = tid >> 5;
    const int wm   = wave >> 2;    // 0..1
    const int wn   = wave & 3;     // 0..3

    const int mBase = blockIdx.y * BM;
    const int nBase = blockIdx.x * BN;

    const int half = lane >> 4;
    const int l16  = lane & 15;
    const int kbA  = half * 16;    // A frag K base (lo VGPRs); +8 for hi VGPRs
    const int kbB  = half * 16;    // B frag K base

    union F16 { v16bf v; v8bf h[2]; };

    v8f acc[4][4];
    const v8f vzero = {0.f,0.f,0.f,0.f,0.f,0.f,0.f,0.f};
#pragma unroll
    for (int i = 0; i < 4; ++i)
#pragma unroll
        for (int j = 0; j < 4; ++j) acc[i][j] = vzero;

    // ---- tile staging: one 16B quad per async op; A:1024 quads, B:2048 quads ----
    auto issue_tiles = [&](int kb, int b) {
#pragma unroll
        for (int it = 0; it < 4; ++it) {           // A tile: 128 rows x 64 k
            int q = tid + it * 256;
            int r = q >> 3, c = q & 7;
            async_copy16(Xh + (size_t)(mBase + r) * IN_F + kb + c * 8,
                         &Asb[b][r][c * 8]);
        }
#pragma unroll
        for (int it = 0; it < 8; ++it) {           // B tile: 256 rows x 64 k
            int q = tid + it * 256;
            int r = q >> 3, c = q & 7;
            async_copy16(Wh + (size_t)(nBase + r) * IN_F + kb + c * 8,
                         &Bsb[b][r][c * 8]);
        }
    };

    int buf = 0;
    issue_tiles(0, 0);

    for (int kb = 0; kb < IN_F; kb += BK) {
        wait_async0();
        __syncthreads();                           // buf is ready for everyone

        int nkb = kb + BK;
        if (nkb < IN_F) issue_tiles(nkb, buf ^ 1); // prefetch next while computing

        // ---- compute on buf: 2 k-steps of 32, 16 WMMAs each ----
#pragma unroll
        for (int s = 0; s < 2; ++s) {
            v16bf bfrag[4];
#pragma unroll
            for (int j = 0; j < 4; ++j) {
                int n = wn * 64 + j * 16 + l16;
                F16 t;
                t.h[0] = *(const v8bf*)&Bsb[buf][n][s * 32 + kbB];
                t.h[1] = *(const v8bf*)&Bsb[buf][n][s * 32 + kbB + 8];
                bfrag[j] = t.v;
            }
#pragma unroll
            for (int i = 0; i < 4; ++i) {
                int m = wm * 64 + i * 16 + l16;
                F16 a;
                a.h[0] = *(const v8bf*)&Asb[buf][m][s * 32 + kbA];
                a.h[1] = *(const v8bf*)&Asb[buf][m][s * 32 + kbA + 8];
#pragma unroll
                for (int j = 0; j < 4; ++j) {
                    acc[i][j] = __builtin_amdgcn_wmma_f32_16x16x32_bf16(
                        false, a.v, false, bfrag[j],
                        (short)0, acc[i][j], false, false);
                }
            }
        }
        buf ^= 1;
    }

    // ---- epilogue: C/D layout (VGPR r -> M = r + 8*half, N = l16) ----
#pragma unroll
    for (int j = 0; j < 4; ++j) {
        int col = nBase + wn * 64 + j * 16 + l16;
        float bv = bias[col];
#pragma unroll
        for (int i = 0; i < 4; ++i) {
            int rowb = mBase + wm * 64 + i * 16 + half * 8;
#pragma unroll
            for (int r = 0; r < 8; ++r) {
                Y[(size_t)(rowb + r) * OUT_F + col] = acc[i][j][r] + bv;
            }
        }
    }
}

// =====================================================================
// Fallback path (small workspace): round-1 kernel, fp32 staging + cvt.
// =====================================================================
__global__ __launch_bounds__(256, 1) void spmm_wmma_kernel(
    const float* __restrict__ X, const float* __restrict__ W,
    const float* __restrict__ bias, float* __restrict__ Y)
{
    __shared__ __align__(16) __bf16 As[FBM][FSTRIDE];
    __shared__ __align__(16) __bf16 Bs[FBN][FSTRIDE];

    const int tid  = threadIdx.x;
    const int lane = tid & 31;
    const int wave = tid >> 5;
    const int wm   = wave >> 2;
    const int wn   = wave & 3;

    const int mBase = blockIdx.y * FBM;
    const int nBase = blockIdx.x * FBN;

    v8f acc[4][2];
    const v8f vzero = {0.f,0.f,0.f,0.f,0.f,0.f,0.f,0.f};
#pragma unroll
    for (int i = 0; i < 4; ++i)
#pragma unroll
        for (int j = 0; j < 2; ++j) acc[i][j] = vzero;

    const int r0 = tid >> 3;
    const int q0 = tid & 7;
    const int half = lane >> 4;
    const int l16  = lane & 15;
    const int kbA  = half * 16;
    const int kbB  = half * 16;

    union F16 { v16bf v; v8bf h[2]; };

    for (int kb = 0; kb < IN_F; kb += FBK) {
#pragma unroll
        for (int it = 0; it < 4; ++it) {
            int r = r0 + it * 32;
            v4f v = *(const v4f*)(X + (size_t)(mBase + r) * IN_F + kb + q0 * 4);
            *(v4bf*)&As[r][q0 * 4] = __builtin_convertvector(v, v4bf);
        }
#pragma unroll
        for (int it = 0; it < 4; ++it) {
            int r = r0 + it * 32;
            v4f v = *(const v4f*)(W + (size_t)(nBase + r) * IN_F + kb + q0 * 4);
            *(v4bf*)&Bs[r][q0 * 4] = __builtin_convertvector(v, v4bf);
        }
        __syncthreads();

        v16bf bfrag[2];
#pragma unroll
        for (int j = 0; j < 2; ++j) {
            int n = wn * 32 + j * 16 + l16;
            F16 b;
            b.h[0] = *(const v8bf*)&Bs[n][kbB];
            b.h[1] = *(const v8bf*)&Bs[n][kbB + 8];
            bfrag[j] = b.v;
        }
#pragma unroll
        for (int i = 0; i < 4; ++i) {
            int m = wm * 64 + i * 16 + l16;
            F16 a;
            a.h[0] = *(const v8bf*)&As[m][kbA];
            a.h[1] = *(const v8bf*)&As[m][kbA + 8];
#pragma unroll
            for (int j = 0; j < 2; ++j) {
                acc[i][j] = __builtin_amdgcn_wmma_f32_16x16x32_bf16(
                    false, a.v, false, bfrag[j], (short)0, acc[i][j], false, false);
            }
        }
        __syncthreads();
    }

#pragma unroll
    for (int j = 0; j < 2; ++j) {
        int col = nBase + wn * 32 + j * 16 + l16;
        float bv = bias[col];
#pragma unroll
        for (int i = 0; i < 4; ++i) {
            int rowb = mBase + wm * 64 + i * 16 + half * 8;
#pragma unroll
            for (int r = 0; r < 8; ++r) {
                Y[(size_t)(rowb + r) * OUT_F + col] = acc[i][j][r] + bv;
            }
        }
    }
}

extern "C" void kernel_launch(void* const* d_in, const int* in_sizes, int n_in,
                              void* d_out, int out_size, void* d_ws, size_t ws_size,
                              hipStream_t stream) {
    const float* x    = (const float*)d_in[0];
    const int*   rows = (const int*)d_in[1];
    const int*   cols = (const int*)d_in[2];
    const float* vals = (const float*)d_in[3];
    const float* bias = (const float*)d_in[4];
    float* Y = (float*)d_out;

    const int nnz = in_sizes[3];

    const size_t W_F32_BYTES = (size_t)OUT_F * IN_F * 4;          // 64 MB
    const size_t W_BF_BYTES  = (size_t)OUT_F * IN_F * 2;          // 32 MB
    const size_t X_BF_BYTES  = (size_t)BATCH * IN_F * 2;          // 64 MB

    float* Wf = (float*)d_ws;                                     // [0, 64MB)

    // 1) zero dense W fp32
    zero_ws_kernel<<<(OUT_F * (size_t)IN_F) / 4 / 256, 256, 0, stream>>>((v4f*)Wf);
    // 2) scatter-add COO
    scatter_coo_kernel<<<(nnz + 255) / 256, 256, 0, stream>>>(rows, cols, vals, Wf, nnz);

    if (ws_size >= W_F32_BYTES + W_BF_BYTES + X_BF_BYTES) {
        __bf16* Wh = (__bf16*)((char*)d_ws + W_F32_BYTES);        // [64MB, 96MB)
        __bf16* Xh = (__bf16*)((char*)d_ws + W_F32_BYTES + W_BF_BYTES); // [96MB, 160MB)

        // 3) convert W and x to bf16 once
        cvt_bf16_kernel<<<(OUT_F * (size_t)IN_F) / 4 / 256, 256, 0, stream>>>(Wf, Wh);
        cvt_bf16_kernel<<<(BATCH * (size_t)IN_F) / 4 / 256, 256, 0, stream>>>(x, Xh);

        // 4) bf16 GEMM with async double-buffered LDS staging
        dim3 grid(OUT_F / BN, BATCH / BM);
        gemm_bf16_async_kernel<<<grid, 256, 0, stream>>>(Xh, Wh, bias, Y);
    } else {
        // fallback: fp32 staging GEMM (converts per-tile)
        dim3 grid(OUT_F / FBN, BATCH / FBM);
        spmm_wmma_kernel<<<grid, 256, 0, stream>>>(x, Wf, bias, Y);
    }
}
